// MSADecoder_50766513439119
// MI455X (gfx1250) — matrix operand order repaired and checked by reading, not compiled, not run on hardware
//
#include <hip/hip_runtime.h>
#include <math.h>

typedef unsigned short ushortT;
typedef __attribute__((ext_vector_type(16))) __bf16 bf16x16;
typedef __attribute__((ext_vector_type(8)))  float  f32x8;

#define B_  1
#define R_  32
#define L_  256
#define T_  8192      // R_*L_
#define D_  768
#define FFN_ 3072
#define H_  12
#define HD_ 64
#define A_  33
#define SEQ_ 512
#define POS_ 128

// ---------- helpers ----------
__device__ __forceinline__ ushortT f2bf(float f) {
    unsigned u = __float_as_uint(f);
    unsigned r = ((u >> 16) & 1u) + 0x7FFFu;
    return (ushortT)((u + r) >> 16);
}
__device__ __forceinline__ float gelu_f(float x) {
    return 0.5f * x * (1.0f + erff(x * 0.70710678118654752f));
}
// A-fragment (16x32 bf16): lane row = l&15; lanes<16 -> K {0..7,16..23}, lanes>=16 -> +8
__device__ __forceinline__ bf16x16 load_frag_a(const ushortT* row, int koff) {
    bf16x16 v;
    ((uint4*)&v)[0] = *(const uint4*)(row + koff);
    ((uint4*)&v)[1] = *(const uint4*)(row + koff + 16);
    return v;
}
// B-fragment from N-major (Bt) storage: lane col = l&15; lanes<16 -> K {0..7,16..23}, lanes>=16 -> +8
__device__ __forceinline__ bf16x16 load_frag_b(const ushortT* row, int koff) {
    bf16x16 v;
    ((uint4*)&v)[0] = *(const uint4*)(row + koff);
    ((uint4*)&v)[1] = *(const uint4*)(row + koff + 16);
    return v;
}
#define WMMA_BF16(a, b, c) \
    __builtin_amdgcn_wmma_f32_16x16x32_bf16(false, (a), false, (b), (short)0, (c), false, false)

// ---------- generic bf16 WMMA GEMM: out[M,N] = act(alpha*(A@Bt^T + bias)) (+resid) ----------
// A: [M,K] bf16 row-major (lda); Bt: [N,K] bf16 (ldb).
// 128x64 tile per block, 128 thr = 4 waves; each wave: 2 m-frags x 4 n-frags = 8 WMMA/chunk.
__global__ __launch_bounds__(128) void k_gemm(
    const ushortT* __restrict__ A, int lda,
    const ushortT* __restrict__ Bt, int ldb,
    const float* __restrict__ bias,
    const float* __restrict__ resid,
    float* __restrict__ outF, ushortT* __restrict__ outB, int ldo,
    int K, float alpha, int flags)
{
    __shared__ __align__(16) ushortT As[128 * 40];
    __shared__ __align__(16) ushortT Bs[64 * 40];
    const int tid = threadIdx.x, w = tid >> 5, l = tid & 31;
    const int m0 = blockIdx.y * 128, n0 = blockIdx.x * 64;
    const int browg = tid >> 1, bhalf = (tid & 1) * 16;   // B-tile loader mapping
    const ushortT* ag = A + (size_t)(m0 + tid) * lda;     // one full 32-wide row per thread
    const ushortT* bg = Bt + (size_t)(n0 + browg) * ldb + bhalf;
    f32x8 acc0[4], acc1[4];
#pragma unroll
    for (int nt = 0; nt < 4; ++nt)
#pragma unroll
        for (int j = 0; j < 8; ++j) { acc0[nt][j] = 0.f; acc1[nt][j] = 0.f; }
    const int koffA = (l & 16) ? 8 : 0;
    const int koffB = (l & 16) ? 8 : 0;
    const ushortT* arow0 = As + (w * 32 + (l & 15)) * 40;
    const ushortT* arow1 = arow0 + 16 * 40;

    for (int kk = 0; kk < K; kk += 32) {
        uint4 a0 = *(const uint4*)(ag);
        uint4 a1 = *(const uint4*)(ag + 8);
        uint4 a2 = *(const uint4*)(ag + 16);
        uint4 a3 = *(const uint4*)(ag + 24);
        uint4 b0 = *(const uint4*)(bg);
        uint4 b1 = *(const uint4*)(bg + 8);
        ag += 32; bg += 32;
        __builtin_prefetch(ag, 0, 1);     // global_prefetch_b8 next tiles
        __builtin_prefetch(bg, 0, 1);
        __syncthreads();
        *(uint4*)(As + tid * 40)      = a0;
        *(uint4*)(As + tid * 40 + 8)  = a1;
        *(uint4*)(As + tid * 40 + 16) = a2;
        *(uint4*)(As + tid * 40 + 24) = a3;
        *(uint4*)(Bs + browg * 40 + bhalf)     = b0;
        *(uint4*)(Bs + browg * 40 + bhalf + 8) = b1;
        __syncthreads();
        bf16x16 af0 = load_frag_a(arow0, koffA);
        bf16x16 af1 = load_frag_a(arow1, koffA);
        bf16x16 bfv[4];
#pragma unroll
        for (int nt = 0; nt < 4; ++nt)
            bfv[nt] = load_frag_b(Bs + (nt * 16 + (l & 15)) * 40, koffB);
#pragma unroll
        for (int nt = 0; nt < 4; ++nt)
            acc0[nt] = WMMA_BF16(af0, bfv[nt], acc0[nt]);
#pragma unroll
        for (int nt = 0; nt < 4; ++nt)
            acc1[nt] = WMMA_BF16(af1, bfv[nt], acc1[nt]);
    }
    const int ml = (l & 16) ? 8 : 0;
#pragma unroll
    for (int mi = 0; mi < 2; ++mi) {
#pragma unroll
        for (int nt = 0; nt < 4; ++nt) {
            int n = n0 + nt * 16 + (l & 15);
            float bv = bias ? bias[n] : 0.f;
#pragma unroll
            for (int i = 0; i < 8; ++i) {
                int m = m0 + w * 32 + mi * 16 + ml + i;
                float v = ((mi ? acc1[nt][i] : acc0[nt][i]) + bv) * alpha;
                if (flags & 1) v = gelu_f(v);
                if (resid) v += resid[(size_t)m * ldo + n];
                if (outF) outF[(size_t)m * ldo + n] = v;
                if (outB) outB[(size_t)m * ldo + n] = f2bf(v);
            }
        }
    }
}

// ---------- row-attention scores: aw[h][i][j] = sum_{r,d} q[r,i,h,d] k[r,j,h,d] ----------
// grid (jtile=4, itile=4, h=12), 128 thr. K = R*HD = 2048, chunked as (r, 32-d halves).
__global__ __launch_bounds__(128) void k_row_scores(
    const ushortT* __restrict__ q, const ushortT* __restrict__ k, float* __restrict__ aw)
{
    __shared__ __align__(16) ushortT As[64 * 40];
    __shared__ __align__(16) ushortT Bs[64 * 40];
    const int tid = threadIdx.x, w = tid >> 5, l = tid & 31;
    const int h = blockIdx.z, i0 = blockIdx.y * 64, j0 = blockIdx.x * 64;
    const int lrow = tid >> 1, lhalf = (tid & 1) * 16;
    f32x8 acc[4];
#pragma unroll
    for (int nt = 0; nt < 4; ++nt)
#pragma unroll
        for (int j = 0; j < 8; ++j) acc[nt][j] = 0.f;
    const int koffA = (l & 16) ? 8 : 0;
    const int koffB = (l & 16) ? 8 : 0;
    const ushortT* arow = As + (w * 16 + (l & 15)) * 40;

    for (int c = 0; c < 64; ++c) {         // 64 chunks of 32 -> K=2048
        int r = c >> 1, dbase = (c & 1) * 32;
        size_t qoff = ((size_t)(r * L_ + i0 + lrow)) * D_ + h * HD_ + dbase + lhalf;
        size_t koff = ((size_t)(r * L_ + j0 + lrow)) * D_ + h * HD_ + dbase + lhalf;
        uint4 a0 = *(const uint4*)(q + qoff), a1 = *(const uint4*)(q + qoff + 8);
        uint4 b0 = *(const uint4*)(k + koff), b1 = *(const uint4*)(k + koff + 8);
        __syncthreads();
        *(uint4*)(As + lrow * 40 + lhalf)     = a0;
        *(uint4*)(As + lrow * 40 + lhalf + 8) = a1;
        *(uint4*)(Bs + lrow * 40 + lhalf)     = b0;
        *(uint4*)(Bs + lrow * 40 + lhalf + 8) = b1;
        __syncthreads();
        bf16x16 af = load_frag_a(arow, koffA);
        bf16x16 bfv[4];
#pragma unroll
        for (int nt = 0; nt < 4; ++nt)
            bfv[nt] = load_frag_b(Bs + (nt * 16 + (l & 15)) * 40, koffB);
#pragma unroll
        for (int nt = 0; nt < 4; ++nt)
            acc[nt] = WMMA_BF16(af, bfv[nt], acc[nt]);
    }
    const int ml = (l & 16) ? 8 : 0;
#pragma unroll
    for (int nt = 0; nt < 4; ++nt)
#pragma unroll
        for (int i = 0; i < 8; ++i)
            aw[(size_t)h * (L_ * L_) + (size_t)(i0 + w * 16 + ml + i) * L_ +
               (j0 + nt * 16 + (l & 15))] = acc[nt][i];
}

// ---------- row-attention ctx: ctx[r,i,h,d] = sum_j P[h,i,j] v[r,j,h,d] ----------
// grid (itile=4, h*32+r=384), 128 thr; Bt = vtrans[h*64+d][r*L+j] (N-major, contiguous in j)
__global__ __launch_bounds__(128) void k_ctx_row(
    const ushortT* __restrict__ P, const ushortT* __restrict__ vt, ushortT* __restrict__ ctx)
{
    __shared__ __align__(16) ushortT As[64 * 40];
    __shared__ __align__(16) ushortT Bs[64 * 40];
    const int tid = threadIdx.x, w = tid >> 5, l = tid & 31;
    const int h = blockIdx.y >> 5, r = blockIdx.y & 31, i0 = blockIdx.x * 64;
    const int lrow = tid >> 1, lhalf = (tid & 1) * 16;
    f32x8 acc[4];
#pragma unroll
    for (int nt = 0; nt < 4; ++nt)
#pragma unroll
        for (int j = 0; j < 8; ++j) acc[nt][j] = 0.f;
    const int koffA = (l & 16) ? 8 : 0;
    const int koffB = (l & 16) ? 8 : 0;
    const ushortT* arow = As + (w * 16 + (l & 15)) * 40;
    const ushortT* Pb = P + (size_t)h * (L_ * L_);

    for (int kk = 0; kk < L_; kk += 32) {
        size_t aoff = (size_t)(i0 + lrow) * L_ + kk + lhalf;
        size_t boff = (size_t)(h * HD_ + lrow) * T_ + r * L_ + kk + lhalf;
        uint4 a0 = *(const uint4*)(Pb + aoff), a1 = *(const uint4*)(Pb + aoff + 8);
        uint4 b0 = *(const uint4*)(vt + boff), b1 = *(const uint4*)(vt + boff + 8);
        __syncthreads();
        *(uint4*)(As + lrow * 40 + lhalf)     = a0;
        *(uint4*)(As + lrow * 40 + lhalf + 8) = a1;
        *(uint4*)(Bs + lrow * 40 + lhalf)     = b0;
        *(uint4*)(Bs + lrow * 40 + lhalf + 8) = b1;
        __syncthreads();
        bf16x16 af = load_frag_a(arow, koffA);
        bf16x16 bfv[4];
#pragma unroll
        for (int nt = 0; nt < 4; ++nt)
            bfv[nt] = load_frag_b(Bs + (nt * 16 + (l & 15)) * 40, koffB);
#pragma unroll
        for (int nt = 0; nt < 4; ++nt)
            acc[nt] = WMMA_BF16(af, bfv[nt], acc[nt]);
    }
    const int ml = (l & 16) ? 8 : 0;
#pragma unroll
    for (int nt = 0; nt < 4; ++nt)
#pragma unroll
        for (int i = 0; i < 8; ++i) {
            int tok = r * L_ + i0 + w * 16 + ml + i;
            ctx[(size_t)tok * D_ + h * HD_ + nt * 16 + (l & 15)] = f2bf(acc[nt][i]);
        }
}

// ---------- col-attention scores: aw[h,c,i,j] = sum_d q[i,c,h,d] k[j,c,h,d] ----------
// grid (c=256, h=12), 128 thr; out 32x32 per block; K=HD=64; fragments direct from global.
__global__ __launch_bounds__(128) void k_col_scores(
    const ushortT* __restrict__ q, const ushortT* __restrict__ k, float* __restrict__ aw)
{
    const int tid = threadIdx.x, w = tid >> 5, l = tid & 31;
    const int c = blockIdx.x, h = blockIdx.y;
    const int mt = w >> 1, nt = w & 1;
    const int koffA = (l & 16) ? 8 : 0;
    const int koffB = (l & 16) ? 8 : 0;
    const int i = mt * 16 + (l & 15), j = nt * 16 + (l & 15);
    f32x8 acc;
#pragma unroll
    for (int z = 0; z < 8; ++z) acc[z] = 0.f;
#pragma unroll
    for (int kc = 0; kc < HD_; kc += 32) {
        bf16x16 af = load_frag_a(q + (size_t)(i * L_ + c) * D_ + h * HD_ + kc, koffA);
        bf16x16 bfv = load_frag_b(k + (size_t)(j * L_ + c) * D_ + h * HD_ + kc, koffB);
        acc = WMMA_BF16(af, bfv, acc);
    }
    const int ml = (l & 16) ? 8 : 0;
#pragma unroll
    for (int z = 0; z < 8; ++z) {
        int iw = mt * 16 + ml + z;
        aw[((size_t)(h * L_ + c) * R_ + iw) * R_ + nt * 16 + (l & 15)] = acc[z];
    }
}

// ---------- col-attention ctx: ctx[i,c,h,d] = sum_j P[h,c,i,j] v[j,c,h,d] ----------
// grid (c=256, h=12), 128 thr; out 32x64; K=32; Bt = vt2[(c*D + h*64 + d)*32 + j]
__global__ __launch_bounds__(128) void k_ctx_col(
    const ushortT* __restrict__ P, const ushortT* __restrict__ vt2, ushortT* __restrict__ ctx)
{
    const int tid = threadIdx.x, w = tid >> 5, l = tid & 31;
    const int c = blockIdx.x, h = blockIdx.y;
    const int mt = w >> 1, ntb = (w & 1) * 2;
    const int koffA = (l & 16) ? 8 : 0;
    const int koffB = (l & 16) ? 8 : 0;
    bf16x16 af = load_frag_a(P + ((size_t)(h * L_ + c)) * (R_ * R_) + (mt * 16 + (l & 15)) * R_, koffA);
    f32x8 acc[2];
    bf16x16 bfv[2];
#pragma unroll
    for (int t = 0; t < 2; ++t) {
#pragma unroll
        for (int z = 0; z < 8; ++z) acc[t][z] = 0.f;
        int d = (ntb + t) * 16 + (l & 15);
        bfv[t] = load_frag_b(vt2 + ((size_t)(c * D_ + h * HD_ + d)) * R_, koffB);
    }
#pragma unroll
    for (int t = 0; t < 2; ++t)
        acc[t] = WMMA_BF16(af, bfv[t], acc[t]);
    const int ml = (l & 16) ? 8 : 0;
#pragma unroll
    for (int t = 0; t < 2; ++t)
#pragma unroll
        for (int z = 0; z < 8; ++z) {
            int tok = (mt * 16 + ml + z) * L_ + c;
            ctx[(size_t)tok * D_ + h * HD_ + (ntb + t) * 16 + (l & 15)] = f2bf(acc[t][z]);
        }
}

// ---------- elementwise / reduction kernels ----------
__global__ void k_transpose_w(const float* __restrict__ W, ushortT* __restrict__ Wt, int K, int N) {
    int idx = blockIdx.x * blockDim.x + threadIdx.x;
    if (idx >= N * K) return;
    int n = idx / K, kk = idx % K;
    Wt[idx] = f2bf(W[(size_t)kk * N + n]);
}
__global__ void k_f2bf_copy(const float* __restrict__ src, ushortT* __restrict__ dst, int n) {
    int idx = blockIdx.x * blockDim.x + threadIdx.x;
    if (idx < n) dst[idx] = f2bf(src[idx]);
}
// rowvec[r][d] = sum_p (sum_k perm[r,k]*pe(k,p)) * Wp[p][d] + decode_b[d]
__global__ void k_rowvec(const float* __restrict__ perm, const float* __restrict__ Wp,
                         const float* __restrict__ bias, float* __restrict__ rowvec) {
    int idx = blockIdx.x * blockDim.x + threadIdx.x;
    if (idx >= R_ * D_) return;
    int r = idx / D_, d = idx % D_;
    float acc = bias[d];
    for (int p = 0; p < POS_; ++p) {
        int pair = p >> 1;
        float div = __expf(-9.2103403720f * (float)pair / 64.0f);
        float pv = 0.f;
        for (int kk = 0; kk < R_; ++kk) {
            float wv = perm[r * R_ + kk];
            if (wv != 0.f) {
                float ang = (float)kk * div;
                pv += wv * ((p & 1) ? __cosf(ang) : __sinf(ang));
            }
        }
        acc += pv * Wp[(size_t)p * D_ + d];
    }
    rowvec[idx] = acc;
}
__global__ void k_combine(const float* __restrict__ xz, const float* __restrict__ rowvec,
                          float* __restrict__ x) {
    int idx = blockIdx.x * blockDim.x + threadIdx.x;
    if (idx >= T_ * D_) return;
    int d = idx % D_, t = idx / D_;
    int r = t >> 8, lc = t & 255;
    x[idx] = xz[(size_t)lc * D_ + d] + rowvec[(size_t)r * D_ + d];
}
__global__ __launch_bounds__(256) void k_ln(
    const float* __restrict__ x, const float* __restrict__ g, const float* __restrict__ b,
    float* __restrict__ outF, ushortT* __restrict__ outB, int dogelu)
{
    __shared__ float red[256];
    const int t = blockIdx.x, tid = threadIdx.x;
    const float* xr = x + (size_t)t * D_;
    float v0 = xr[tid], v1 = xr[tid + 256], v2 = xr[tid + 512];
    red[tid] = v0 + v1 + v2;
    __syncthreads();
    for (int o = 128; o > 0; o >>= 1) { if (tid < o) red[tid] += red[tid + o]; __syncthreads(); }
    float mu = red[0] * (1.f / 768.f);
    __syncthreads();
    float d0 = v0 - mu, d1 = v1 - mu, d2 = v2 - mu;
    red[tid] = d0 * d0 + d1 * d1 + d2 * d2;
    __syncthreads();
    for (int o = 128; o > 0; o >>= 1) { if (tid < o) red[tid] += red[tid + o]; __syncthreads(); }
    float rstd = rsqrtf(red[0] * (1.f / 768.f) + 1e-5f);
    float vv[3] = { d0, d1, d2 };
#pragma unroll
    for (int j = 0; j < 3; ++j) {
        int d = tid + j * 256;
        float y = vv[j] * rstd * g[d] + b[d];
        if (dogelu) y = gelu_f(y);
        if (outF) outF[(size_t)t * D_ + d] = y;
        if (outB) outB[(size_t)t * D_ + d] = f2bf(y);
    }
}
__global__ __launch_bounds__(256) void k_softmax256(const float* __restrict__ aw, ushortT* __restrict__ P) {
    __shared__ float red[256];
    const int tid = threadIdx.x;
    const size_t base = (size_t)blockIdx.x * 256;
    float v = aw[base + tid];
    red[tid] = v;
    __syncthreads();
    for (int o = 128; o > 0; o >>= 1) { if (tid < o) red[tid] = fmaxf(red[tid], red[tid + o]); __syncthreads(); }
    float mx = red[0];
    __syncthreads();
    float e = __expf(v - mx);
    red[tid] = e;
    __syncthreads();
    for (int o = 128; o > 0; o >>= 1) { if (tid < o) red[tid] += red[tid + o]; __syncthreads(); }
    P[base + tid] = f2bf(e / red[0]);
}
__global__ __launch_bounds__(256) void k_softmax32(const float* __restrict__ aw, ushortT* __restrict__ P) {
    const int wv = threadIdx.x >> 5, l = threadIdx.x & 31;
    const size_t row = (size_t)blockIdx.x * 8 + wv;
    float v = aw[row * 32 + l];
    float m = v;
    for (int o = 16; o > 0; o >>= 1) m = fmaxf(m, __shfl_xor(m, o, 32));
    float e = __expf(v - m);
    float s = e;
    for (int o = 16; o > 0; o >>= 1) s += __shfl_xor(s, o, 32);
    P[row * 32 + l] = f2bf(e / s);
}
__global__ void k_vt(const ushortT* __restrict__ vb, ushortT* __restrict__ vt) {
    int idx = blockIdx.x * blockDim.x + threadIdx.x;   // f*T + t
    if (idx >= D_ * T_) return;
    int f = idx / T_, t = idx % T_;
    vt[idx] = vb[(size_t)t * D_ + f];
}
__global__ void k_vt2(const ushortT* __restrict__ vb, ushortT* __restrict__ vt2) {
    int idx = blockIdx.x * blockDim.x + threadIdx.x;   // (c*D + f)*R + j
    if (idx >= L_ * D_ * R_) return;
    int j = idx & 31;
    int cf = idx >> 5;
    int f = cf % D_, c = cf / D_;
    vt2[idx] = vb[((size_t)j * L_ + c) * D_ + f];
}
__global__ void k_logits(const float* __restrict__ hv, const float* __restrict__ W,
                         const float* __restrict__ b, float* __restrict__ out) {
    int idx = blockIdx.x * blockDim.x + threadIdx.x;
    if (idx >= T_ * A_) return;
    int t = idx / A_, a = idx % A_;
    const float* hr = hv + (size_t)t * D_;
    float acc = b[a];
    for (int kk = 0; kk < D_; ++kk) acc += hr[kk] * W[(size_t)kk * A_ + a];
    out[idx] = acc;
}

// ---------- host driver ----------
extern "C" void kernel_launch(void* const* d_in, const int* in_sizes, int n_in,
                              void* d_out, int out_size, void* d_ws, size_t ws_size,
                              hipStream_t stream) {
    const float* z        = (const float*)d_in[0];
    const float* perm     = (const float*)d_in[1];
    const float* dec_W    = (const float*)d_in[2];
    const float* dec_b    = (const float*)d_in[3];
    const float* before_g = (const float*)d_in[4];
    const float* before_b = (const float*)d_in[5];
    const float* after_g  = (const float*)d_in[6];
    const float* after_b  = (const float*)d_in[7];
    const float* logits_W = (const float*)d_in[8];
    const float* logits_b = (const float*)d_in[9];
    const float* row_ln_g = (const float*)d_in[10];
    const float* row_ln_b = (const float*)d_in[11];
    const float* rqkvoW[4] = { (const float*)d_in[12], (const float*)d_in[14],
                               (const float*)d_in[16], (const float*)d_in[18] };
    const float* rqkvob[4] = { (const float*)d_in[13], (const float*)d_in[15],
                               (const float*)d_in[17], (const float*)d_in[19] };
    const float* col_ln_g = (const float*)d_in[20];
    const float* col_ln_b = (const float*)d_in[21];
    const float* cqkvoW[4] = { (const float*)d_in[22], (const float*)d_in[24],
                               (const float*)d_in[26], (const float*)d_in[28] };
    const float* cqkvob[4] = { (const float*)d_in[23], (const float*)d_in[25],
                               (const float*)d_in[27], (const float*)d_in[29] };
    const float* ffn_ln_g = (const float*)d_in[30];
    const float* ffn_ln_b = (const float*)d_in[31];
    const float* fc1_W = (const float*)d_in[32];
    const float* fc1_b = (const float*)d_in[33];
    const float* fc2_W = (const float*)d_in[34];
    const float* fc2_b = (const float*)d_in[35];
    float* out = (float*)d_out;

    char* wsb = (char*)d_ws;
    size_t off = 0;
    auto carve = [&](size_t bytes) -> char* {
        char* p = wsb + off;
        off = (off + bytes + 255) & ~(size_t)255;
        return p;
    };
    ushortT* wzT   = (ushortT*)carve((size_t)D_ * SEQ_ * 2);
    ushortT* wqkvoT= (ushortT*)carve((size_t)16 * D_ * D_ * 2);      // [layer][pre][mat]
    ushortT* wfc1T = (ushortT*)carve((size_t)2 * FFN_ * D_ * 2);
    ushortT* wfc2T = (ushortT*)carve((size_t)2 * D_ * FFN_ * 2);
    ushortT* zb    = (ushortT*)carve((size_t)L_ * SEQ_ * 2);
    float*   xzf   = (float*)  carve((size_t)L_ * D_ * 4);
    float*   rowv  = (float*)  carve((size_t)R_ * D_ * 4);
    float*   x     = (float*)  carve((size_t)T_ * D_ * 4);
    ushortT* hbf   = (ushortT*)carve((size_t)T_ * D_ * 2);
    ushortT* qb    = (ushortT*)carve((size_t)T_ * D_ * 2);
    ushortT* kb    = (ushortT*)carve((size_t)T_ * D_ * 2);
    ushortT* vb    = (ushortT*)carve((size_t)T_ * D_ * 2);
    ushortT* vt    = (ushortT*)carve((size_t)D_ * T_ * 2);
    ushortT* vt2   = (ushortT*)carve((size_t)L_ * D_ * R_ * 2);
    float*   aw    = (float*)  carve((size_t)H_ * L_ * R_ * R_ * 4); // >= row aw size too
    ushortT* Pb    = (ushortT*)carve((size_t)H_ * L_ * R_ * R_ * 2);
    ushortT* ctxb  = (ushortT*)carve((size_t)T_ * D_ * 2);
    ushortT* ub    = (ushortT*)carve((size_t)T_ * FFN_ * 2);
    float*   hv    = (float*)  carve((size_t)T_ * D_ * 4);
    (void)ws_size; (void)n_in; (void)in_sizes; (void)out_size;

    auto tw = [&](const float* W, ushortT* Wt, int K, int N) {
        int n = N * K;
        k_transpose_w<<<(n + 255) / 256, 256, 0, stream>>>(W, Wt, K, N);
    };
    auto gemm = [&](const ushortT* A, int lda, const ushortT* Bt, int ldb,
                    const float* bias, const float* resid,
                    float* outF, ushortT* outB, int ldo,
                    int M, int N, int K, float alpha, int flags) {
        dim3 g(N / 64, M / 128);
        k_gemm<<<g, 128, 0, stream>>>(A, lda, Bt, ldb, bias, resid, outF, outB, ldo, K, alpha, flags);
    };

    // ---- one-time weight conversion (deterministic; rerun every call) ----
    tw(dec_W, wzT, SEQ_, D_);                       // first 512 rows of decode_z_W
    for (int lyr = 0; lyr < 2; ++lyr) {
        for (int m = 0; m < 4; ++m) {
            tw(rqkvoW[m] + (size_t)lyr * D_ * D_, wqkvoT + ((size_t)(lyr * 2 + 0) * 4 + m) * D_ * D_, D_, D_);
            tw(cqkvoW[m] + (size_t)lyr * D_ * D_, wqkvoT + ((size_t)(lyr * 2 + 1) * 4 + m) * D_ * D_, D_, D_);
        }
        tw(fc1_W + (size_t)lyr * D_ * FFN_, wfc1T + (size_t)lyr * FFN_ * D_, D_, FFN_);
        tw(fc2_W + (size_t)lyr * FFN_ * D_, wfc2T + (size_t)lyr * D_ * FFN_, FFN_, D_);
    }

    // ---- embedding ----
    k_f2bf_copy<<<(L_ * SEQ_ + 255) / 256, 256, 0, stream>>>(z, zb, L_ * SEQ_);
    k_rowvec<<<(R_ * D_ + 255) / 256, 256, 0, stream>>>(perm, dec_W + (size_t)SEQ_ * D_, dec_b, rowv);
    gemm(zb, SEQ_, wzT, SEQ_, nullptr, nullptr, xzf, nullptr, D_, L_, D_, SEQ_, 1.f, 0);
    k_combine<<<(T_ * D_ + 255) / 256, 256, 0, stream>>>(xzf, rowv, x);
    k_ln<<<T_, 256, 0, stream>>>(x, before_g, before_b, x, nullptr, 0);

    const float row_scale = 1.0f / (8.0f * sqrtf(32.0f));  // HD^-0.5 / sqrt(R)
    const float col_scale = 0.125f;                        // HD^-0.5

    for (int lyr = 0; lyr < 2; ++lyr) {
        ushortT* rowW = wqkvoT + (size_t)(lyr * 2 + 0) * 4 * D_ * D_;
        ushortT* colW = wqkvoT + (size_t)(lyr * 2 + 1) * 4 * D_ * D_;

        // ===== row self-attention (tied over rows) =====
        k_ln<<<T_, 256, 0, stream>>>(x, row_ln_g + lyr * D_, row_ln_b + lyr * D_, nullptr, hbf, 0);
        gemm(hbf, D_, rowW + 0 * D_ * D_, D_, rqkvob[0] + lyr * D_, nullptr, nullptr, qb, D_, T_, D_, D_, row_scale, 0);
        gemm(hbf, D_, rowW + 1 * D_ * D_, D_, rqkvob[1] + lyr * D_, nullptr, nullptr, kb, D_, T_, D_, D_, 1.f, 0);
        gemm(hbf, D_, rowW + 2 * D_ * D_, D_, rqkvob[2] + lyr * D_, nullptr, nullptr, vb, D_, T_, D_, D_, 1.f, 0);
        k_vt<<<(D_ * T_ + 255) / 256, 256, 0, stream>>>(vb, vt);
        k_row_scores<<<dim3(4, 4, H_), 128, 0, stream>>>(qb, kb, aw);
        k_softmax256<<<H_ * L_, 256, 0, stream>>>(aw, Pb);
        k_ctx_row<<<dim3(4, H_ * R_), 128, 0, stream>>>(Pb, vt, ctxb);
        gemm(ctxb, D_, rowW + 3 * D_ * D_, D_, rqkvob[3] + lyr * D_, x, x, nullptr, D_, T_, D_, D_, 1.f, 0);

        // ===== column self-attention =====
        k_ln<<<T_, 256, 0, stream>>>(x, col_ln_g + lyr * D_, col_ln_b + lyr * D_, nullptr, hbf, 0);
        gemm(hbf, D_, colW + 0 * D_ * D_, D_, cqkvob[0] + lyr * D_, nullptr, nullptr, qb, D_, T_, D_, D_, col_scale, 0);
        gemm(hbf, D_, colW + 1 * D_ * D_, D_, cqkvob[1] + lyr * D_, nullptr, nullptr, kb, D_, T_, D_, D_, 1.f, 0);
        gemm(hbf, D_, colW + 2 * D_ * D_, D_, cqkvob[2] + lyr * D_, nullptr, nullptr, vb, D_, T_, D_, D_, 1.f, 0);
        k_vt2<<<(L_ * D_ * R_ + 255) / 256, 256, 0, stream>>>(vb, vt2);
        k_col_scores<<<dim3(L_, H_), 128, 0, stream>>>(qb, kb, aw);
        k_softmax32<<<(H_ * L_ * R_) / 8, 256, 0, stream>>>(aw, Pb);
        k_ctx_col<<<dim3(L_, H_), 128, 0, stream>>>(Pb, vt2, ctxb);
        gemm(ctxb, D_, colW + 3 * D_ * D_, D_, cqkvob[3] + lyr * D_, x, x, nullptr, D_, T_, D_, D_, 1.f, 0);

        // ===== FFN =====
        k_ln<<<T_, 256, 0, stream>>>(x, ffn_ln_g + lyr * D_, ffn_ln_b + lyr * D_, nullptr, hbf, 0);
        gemm(hbf, D_, wfc1T + (size_t)lyr * FFN_ * D_, D_, fc1_b + lyr * FFN_, nullptr,
             nullptr, ub, FFN_, T_, FFN_, D_, 1.f, /*gelu*/1);
        gemm(ub, FFN_, wfc2T + (size_t)lyr * D_ * FFN_, FFN_, fc2_b + lyr * D_, x,
             x, nullptr, D_, T_, D_, FFN_, 1.f, 0);
    }

    // ---- final LN + gelu + logits ----
    k_ln<<<T_, 256, 0, stream>>>(x, after_g, after_b, hv, nullptr, /*gelu*/1);
    k_logits<<<(T_ * A_ + 255) / 256, 256, 0, stream>>>(hv, logits_W, logits_b, out);
}
